// ScaleNet_81166291960011
// MI455X (gfx1250) — matrix operand order repaired and checked by reading, not compiled, not run on hardware
//
#include <hip/hip_runtime.h>
#include <stdint.h>

#define NB 32
#define CC 256
#define HH 56
#define WW 56
#define HW (HH * WW)           // 3136
#define PP (NB * HW)           // 100352 pixels
#define KSTEPS 72              // 9 taps * 8 ci-blocks of 32
#define BN_EPS 1e-5f

typedef __attribute__((ext_vector_type(16))) __bf16 v16bf;
typedef __attribute__((ext_vector_type(8)))  float  v8f;
typedef __attribute__((ext_vector_type(4)))  int    v4i;

union FragU { uint4 u[2]; v16bf v; };

#define AS1 __attribute__((address_space(1)))
#define AS3 __attribute__((address_space(3)))

#if __has_builtin(__builtin_amdgcn_global_load_async_to_lds_b128) && \
    __has_builtin(__builtin_amdgcn_s_wait_asynccnt)
#define USE_ASYNC 1
#else
#define USE_ASYNC 0
#endif

__device__ __forceinline__ unsigned short f2bf(float f) {
    union { float f; unsigned int u; } x; x.f = f;
    unsigned int r = x.u + 0x7FFFu + ((x.u >> 16) & 1u);   // round-to-nearest-even
    return (unsigned short)(r >> 16);
}

// ---- x (NCHW fp32) -> NHWC bf16 -------------------------------------------
__global__ __launch_bounds__(256) void k_cvt_nhwc(const float* __restrict__ x,
                                                  unsigned short* __restrict__ xb) {
    int o = blockIdx.x * 256 + threadIdx.x;
    if (o >= PP * CC) return;
    int c = o & 255;
    int p = o >> 8;
    int n = p / HW; int rem = p - n * HW;
    xb[o] = f2bf(x[((size_t)(n * CC + c)) * HW + rem]);
}

// ---- w (OIHW fp32) -> wt[tap][cib][co][ci32] bf16 (B-tile ready layout) ----
__global__ __launch_bounds__(256) void k_wre(const float* __restrict__ w,
                                             unsigned short* __restrict__ wt) {
    int o = blockIdx.x * 256 + threadIdx.x;     // ((r*8+cib)*256+co)*32+cil
    if (o >= 9 * 8 * 256 * 32) return;
    int cil = o & 31; int t = o >> 5;
    int co  = t & 255; t >>= 8;
    int cib = t & 7;   int r = t >> 3;
    int ci  = cib * 32 + cil;
    wt[o] = f2bf(w[(co * CC + ci) * 9 + r]);
}

// ---- hard Gumbel-sigmoid gates: sigmoid(conv(x)+b) > 0.5  <=>  z > 0 ------
__global__ __launch_bounds__(128) void k_mask(const float* __restrict__ x,
    const float* __restrict__ wm1, const float* __restrict__ bm1,
    const float* __restrict__ wm2, const float* __restrict__ bm2,
    float* __restrict__ m1, float* __restrict__ m2) {
    int p = blockIdx.x * 128 + threadIdx.x;
    if (p >= PP) return;
    int w = p % WW; int t = p / WW; int h = t % HH; int n = t / HH;
    float s1 = bm1[0], s2 = bm2[0];
    for (int ci = 0; ci < CC; ++ci) {
        const float* xc = x + ((size_t)(n * CC + ci)) * HW;
        const float* wA = wm1 + ci * 9;
        const float* wB = wm2 + ci * 9;
#pragma unroll
        for (int r = 0; r < 9; ++r) {
            int hh = h + r / 3 - 1, ww = w + r % 3 - 1;
            if (hh >= 0 && hh < HH && ww >= 0 && ww < WW) {
                float xv = xc[hh * WW + ww];
                s1 = fmaf(xv, wA[r], s1);
                s2 = fmaf(xv, wB[r], s2);
            }
        }
    }
    m1[p] = s1 > 0.f ? 1.f : 0.f;
    m2[p] = s2 > 0.f ? 1.f : 0.f;
}

// ---- implicit-GEMM 3x3 conv, bf16 WMMA, fused BN (+relu/mask/residual) ----
// Block tile: 128 pixels x 64 out-channels; 8 waves as 4(M) x 2(N);
// each wave: 32x32 via 2x2 WMMA 16x16x32 bf16 accumulators.
// A and B tiles are pure async global->LDS b128 copies (ASYNCcnt); loads for
// step kk+1 are issued BEFORE the WMMAs of step kk so the Tensor/async engine
// runs under the matrix math (true double buffering).
// EPI==0: out = bf16( relu(bn(acc)) * mask )            -> NHWC bf16 (direct)
// EPI==1: out = relu( bn(acc)*mask + x_res )            -> NCHW fp32 via LDS
//         transpose so residual reads + stores are fully coalesced.
template <int EPI>
__global__ __launch_bounds__(256) void k_conv_wmma(
    const unsigned short* __restrict__ xin,   // NHWC bf16
    const unsigned short* __restrict__ wt,    // [9][8][256][32] bf16
    const float* __restrict__ gg, const float* __restrict__ bb,
    const float* __restrict__ mm, const float* __restrict__ vv,
    const float* __restrict__ mask,
    const float* __restrict__ xres,
    unsigned short* __restrict__ obf,
    float* __restrict__ of32)
{
    __shared__ union SM {
        struct {
            unsigned short As[2][128][40];  // [pixel][k]  (+pad, 16B-aligned rows)
            unsigned short Bt[2][64][40];   // [co_local][k]
        } s;
        float T[128][65];                   // fp32 staging for coalesced epilogue
    } sm;

    const int tid    = threadIdx.x;
    const int ptile  = blockIdx.x * 128;
    const int cobase = blockIdx.y * 64;

    // A-tile loader: 128 pixels x 32 k; 32B (2 x b128) per thread
    const int a_pix = tid >> 1;
    const int a_cio = (tid & 1) * 16;
    int ap = ptile + a_pix;
    const int aw = ap % WW; int t0 = ap / WW;
    const int ah = t0 % HH; const int an = t0 / HH;

    // B-tile loader: 64 co x 32 ci; 16B (1 x b128) per thread
    const int b_co  = tid >> 2;
    const int b_cio = (tid & 3) * 8;

    const int wave  = tid >> 5;
    const int lane  = tid & 31;
    const int wm_   = wave & 3;       // M sub-tile (4 x 32 pixels)
    const int wn_   = wave >> 2;      // N sub-tile (2 x 32 channels)
    const int row16 = lane & 15;
    const int hi    = lane >> 4;      // lane half selects K sub-range

    v8f acc[2][2] = {{{}, {}}, {{}, {}}};

    auto load_tile = [&](int kk, int buf) {
        const int r   = kk >> 3;                  // tap 0..8
        const int cib = kk & 7;                   // ci block 0..7
        const int dh  = r / 3 - 1, dw = r % 3 - 1;
        // A: 16 contiguous channels of one shifted pixel (zero-pad edges)
        const int hh = ah + dh, ww2 = aw + dw;
        unsigned short* ldst = &sm.s.As[buf][a_pix][a_cio];
        if (hh >= 0 && hh < HH && ww2 >= 0 && ww2 < WW) {
            const unsigned short* gsrc =
                xin + ((size_t)((an * HH + hh) * WW + ww2) * CC + cib * 32 + a_cio);
#if USE_ASYNC
            __builtin_amdgcn_global_load_async_to_lds_b128(
                (AS1 v4i*)gsrc, (AS3 v4i*)ldst, 0, 0);
            __builtin_amdgcn_global_load_async_to_lds_b128(
                (AS1 v4i*)(gsrc + 8), (AS3 v4i*)(ldst + 8), 0, 0);
#else
            ((uint4*)ldst)[0] = ((const uint4*)gsrc)[0];
            ((uint4*)ldst)[1] = ((const uint4*)gsrc)[1];
#endif
        } else {
            uint4 z = make_uint4(0u, 0u, 0u, 0u);
            ((uint4*)ldst)[0] = z;
            ((uint4*)ldst)[1] = z;
        }
        // B: weights are pre-transposed; tile is a pure 16B/thread copy
        const unsigned short* bsrc =
            wt + ((size_t)((r * 8 + cib) * 256 + cobase + b_co) * 32 + b_cio);
        unsigned short* bdst = &sm.s.Bt[buf][b_co][b_cio];
        __builtin_prefetch(bsrc + 256 * 32, 0, 0);   // next ci-block tile
#if USE_ASYNC
        __builtin_amdgcn_global_load_async_to_lds_b128(
            (AS1 v4i*)bsrc, (AS3 v4i*)bdst, 0, 0);
#else
        *(uint4*)bdst = *(const uint4*)bsrc;
#endif
    };

    load_tile(0, 0);

    for (int kk = 0; kk < KSTEPS; ++kk) {
#if USE_ASYNC
        __builtin_amdgcn_s_wait_asynccnt(0);   // buf[kk&1] resident
#endif
        __syncthreads();                       // all waves done reading buf^1
        const int buf = kk & 1;

        // Fragments per ISA VGPR layouts (wave32):
        // A 16x32: lanes 0-15 K[0..7],K[16..23]; lanes 16-31 K[8..15],K[24..31]
        // B 32x16: lanes 0-15 K[0..7],K[16..23]; lanes 16-31 K[8..15],K[24..31] (row = column N)
        FragU fa[2], fb[2];
        const int ar0 = 32 * wm_ + row16;
#pragma unroll
        for (int mt = 0; mt < 2; ++mt) {
            fa[mt].u[0] = *(const uint4*)&sm.s.As[buf][ar0 + 16 * mt][hi * 8];
            fa[mt].u[1] = *(const uint4*)&sm.s.As[buf][ar0 + 16 * mt][16 + hi * 8];
        }
        const int br0 = 32 * wn_ + row16;
#pragma unroll
        for (int nt = 0; nt < 2; ++nt) {
            fb[nt].u[0] = *(const uint4*)&sm.s.Bt[buf][br0 + 16 * nt][hi * 8];
            fb[nt].u[1] = *(const uint4*)&sm.s.Bt[buf][br0 + 16 * nt][16 + hi * 8];
        }

        // Issue next tile's async copies BEFORE the WMMAs so they overlap.
        if (kk + 1 < KSTEPS) load_tile(kk + 1, buf ^ 1);

#pragma unroll
        for (int mt = 0; mt < 2; ++mt)
#pragma unroll
            for (int nt = 0; nt < 2; ++nt)
                acc[mt][nt] = __builtin_amdgcn_wmma_f32_16x16x32_bf16(
                    false, fa[mt].v, false, fb[nt].v, (short)0, acc[mt][nt],
                    false, false);
    }

    // ---------------- Epilogue ----------------
    if (EPI == 0) {
        // bf16 NHWC: lanes of a wave cover 16 consecutive co -> contiguous stores
#pragma unroll
        for (int nt = 0; nt < 2; ++nt) {
            const int co = cobase + 32 * wn_ + 16 * nt + row16;
            const float s = gg[co] * rsqrtf(vv[co] + BN_EPS);
            const float tt = bb[co] - mm[co] * s;
#pragma unroll
            for (int mt = 0; mt < 2; ++mt)
#pragma unroll
                for (int j = 0; j < 8; ++j) {
                    const int pp2 = ptile + 32 * wm_ + 16 * mt + j + 8 * hi;
                    float val = fmaxf(acc[mt][nt][j] * s + tt, 0.f) * mask[pp2];
                    obf[(size_t)pp2 * CC + co] = f2bf(val);
                }
        }
    } else {
        // Stage bn(acc) in LDS, then coalesced mask+residual+relu+store (NCHW).
        __syncthreads();   // done reading As/Bt; safe to overlay T
#pragma unroll
        for (int nt = 0; nt < 2; ++nt) {
            const int co = cobase + 32 * wn_ + 16 * nt + row16;
            const float s = gg[co] * rsqrtf(vv[co] + BN_EPS);
            const float tt = bb[co] - mm[co] * s;
            const int co_l = 32 * wn_ + 16 * nt + row16;
#pragma unroll
            for (int mt = 0; mt < 2; ++mt)
#pragma unroll
                for (int j = 0; j < 8; ++j) {
                    const int pl = 32 * wm_ + 16 * mt + j + 8 * hi;
                    sm.T[pl][co_l] = acc[mt][nt][j] * s + tt;
                }
        }
        __syncthreads();
        // 128 pixels x 64 co; consecutive threads -> consecutive pixels, one co
#pragma unroll 4
        for (int i = 0; i < 32; ++i) {
            const int idx  = i * 256 + tid;
            const int pl   = idx & 127;
            const int co_l = idx >> 7;
            const int p    = ptile + pl;
            const int n    = p / HW;
            const int rem  = p - n * HW;
            const size_t gidx = (size_t)(n * CC + cobase + co_l) * HW + rem;
            float val = sm.T[pl][co_l] * mask[p] + xres[gidx];
            of32[gidx] = fmaxf(val, 0.f);
        }
    }
}

extern "C" void kernel_launch(void* const* d_in, const int* in_sizes, int n_in,
                              void* d_out, int out_size, void* d_ws, size_t ws_size,
                              hipStream_t stream) {
    const float* x   = (const float*)d_in[0];
    const float* w1  = (const float*)d_in[1];
    const float* w2  = (const float*)d_in[2];
    const float* g1  = (const float*)d_in[3];
    const float* b1  = (const float*)d_in[4];
    const float* mu1 = (const float*)d_in[5];
    const float* v1  = (const float*)d_in[6];
    const float* g2  = (const float*)d_in[7];
    const float* b2  = (const float*)d_in[8];
    const float* mu2 = (const float*)d_in[9];
    const float* v2  = (const float*)d_in[10];
    const float* wm1 = (const float*)d_in[11];
    const float* bm1 = (const float*)d_in[12];
    const float* wm2 = (const float*)d_in[13];
    const float* bm2 = (const float*)d_in[14];
    float* out = (float*)d_out;

    char* ws = (char*)d_ws;
    unsigned short* xb  = (unsigned short*)ws; ws += (size_t)PP * CC * 2;  // x NHWC bf16
    unsigned short* ob  = (unsigned short*)ws; ws += (size_t)PP * CC * 2;  // stage1 NHWC bf16
    unsigned short* wt1 = (unsigned short*)ws; ws += (size_t)9 * CC * CC * 2;
    unsigned short* wt2 = (unsigned short*)ws; ws += (size_t)9 * CC * CC * 2;
    float* m1 = (float*)ws; ws += (size_t)PP * 4;
    float* m2 = (float*)ws; ws += (size_t)PP * 4;

    k_cvt_nhwc<<<(PP * CC + 255) / 256, 256, 0, stream>>>(x, xb);
    k_wre<<<(9 * CC * CC + 255) / 256, 256, 0, stream>>>(w1, wt1);
    k_wre<<<(9 * CC * CC + 255) / 256, 256, 0, stream>>>(w2, wt2);
    k_mask<<<(PP + 127) / 128, 128, 0, stream>>>(x, wm1, bm1, wm2, bm2, m1, m2);

    dim3 grid(PP / 128, CC / 64);   // 784 x 4 workgroups
    k_conv_wmma<0><<<grid, 256, 0, stream>>>(xb, wt1, g1, b1, mu1, v1, m1,
                                             nullptr, ob, nullptr);
    k_conv_wmma<1><<<grid, 256, 0, stream>>>(ob, wt2, g2, b2, mu2, v2, m2,
                                             x, nullptr, out);
}
